// GraphKHopDecentDiffAttentionLayer_50938312131106
// MI455X (gfx1250) — compile-verified, matrix operand order checked by reading, not run on hardware
//
#include <hip/hip_runtime.h>
#include <hip/hip_bf16.h>

#define N_NODES 4096
#define IN_F    256
#define OUT_F   128
#define K_HOPS  3
#define BETA    0.5f
#define ALPHA   0.2f
#define NEG_BIG (-9.0e15f)

typedef __attribute__((ext_vector_type(16))) __bf16       v16bf;
typedef __attribute__((ext_vector_type(8)))  float        v8f;
typedef __attribute__((ext_vector_type(4)))  float        v4f;
typedef __attribute__((ext_vector_type(8)))  unsigned int v8u;
typedef __attribute__((ext_vector_type(4)))  unsigned int v4u;

__device__ __forceinline__ unsigned short f2bf(float x) {
    unsigned int u = __builtin_bit_cast(unsigned int, x);
    u += 0x7FFFu + ((u >> 16) & 1u);   // round-to-nearest-even
    return (unsigned short)(u >> 16);
}

// ---------------------------------------------------------------------------
// Kernel 1: Whnew_i = h @ W_i (fp32), f1_i = Whnew_i@a1, f2_i = Whnew_i@a2
// ---------------------------------------------------------------------------
__global__ void __launch_bounds__(256)
k_proj(const float* __restrict__ h, const float* __restrict__ W,
       const float* __restrict__ a, float* __restrict__ whF,
       float* __restrict__ f1, float* __restrict__ f2)
{
    __shared__ float hl[16][IN_F];       // 16 KB
    __shared__ float Ws[64][OUT_F];      // 32 KB
    __shared__ float red1[256], red2[256];
    const int t = threadIdx.x;
    const int nbase = blockIdx.x * 16;

    for (int idx = t; idx < 16 * IN_F; idx += 256) {
        int node = idx >> 8, k = idx & 255;
        hl[node][k] = h[(size_t)(nbase + node) * IN_F + k];
    }

    const int node  = t >> 4;
    const int gnode = nbase + node;
    const int oc    = (t & 15) * 8;
    const int wk    = t >> 2;
    const int wc    = (t & 3) * 32;

    for (int i = 0; i < K_HOPS; ++i) {
        float acc[8];
#pragma unroll
        for (int j = 0; j < 8; ++j) acc[j] = 0.f;

        for (int kb = 0; kb < IN_F; kb += 64) {
            __syncthreads();
            const float4* wsrc = reinterpret_cast<const float4*>(
                W + ((size_t)i * IN_F + kb + wk) * OUT_F + wc);
            float4* wdst = reinterpret_cast<float4*>(&Ws[wk][wc]);
#pragma unroll
            for (int q = 0; q < 8; ++q) wdst[q] = wsrc[q];
            __syncthreads();
            for (int k2 = 0; k2 < 64; ++k2) {
                float hv = hl[node][kb + k2];
                const float4* wr = reinterpret_cast<const float4*>(&Ws[k2][oc]);
                float4 w0 = wr[0], w1 = wr[1];
                acc[0] += hv * w0.x; acc[1] += hv * w0.y;
                acc[2] += hv * w0.z; acc[3] += hv * w0.w;
                acc[4] += hv * w1.x; acc[5] += hv * w1.y;
                acc[6] += hv * w1.z; acc[7] += hv * w1.w;
            }
        }

        size_t base = ((size_t)i * N_NODES + gnode) * OUT_F + oc;
        const float* a1 = a + (size_t)(2 * i) * OUT_F;
        const float* a2 = a + (size_t)(2 * i + 1) * OUT_F;
        float p1 = 0.f, p2 = 0.f;
#pragma unroll
        for (int j = 0; j < 8; ++j) {
            whF[base + j] = acc[j];
            p1 += acc[j] * a1[oc + j];
            p2 += acc[j] * a2[oc + j];
        }
        red1[t] = p1; red2[t] = p2;
        __syncthreads();
        for (int off = 8; off >= 1; off >>= 1) {
            if ((t & 15) < off) { red1[t] += red1[t + off]; red2[t] += red2[t + off]; }
            __syncthreads();
        }
        if ((t & 15) == 0) {
            f1[(size_t)i * N_NODES + gnode] = red1[t];
            f2[(size_t)i * N_NODES + gnode] = red2[t];
        }
        __syncthreads();
    }
}

// ---------------------------------------------------------------------------
// Kernel 2: per hop: f2max, plus column exp tables
//   e1[c] = exp(f2[c]-f2max)        (<=1)
//   e2[c] = exp(ALPHA*(f2[c]-f2max))(<=1)
// Removes the per-mask-element transcendental from the attention kernel.
// ---------------------------------------------------------------------------
__global__ void __launch_bounds__(256)
k_fprep(const float* __restrict__ f2, float* __restrict__ fmax,
        float* __restrict__ e1, float* __restrict__ e2)
{
    __shared__ float sm[256];
    const int i = blockIdx.x, t = threadIdx.x;
    const float* v = f2 + (size_t)i * N_NODES;
    float m = -INFINITY;
    for (int c = t; c < N_NODES; c += 256) m = fmaxf(m, v[c]);
    sm[t] = m;
    __syncthreads();
    for (int off = 128; off >= 1; off >>= 1) {
        if (t < off) sm[t] = fmaxf(sm[t], sm[t + off]);
        __syncthreads();
    }
    const float fm = sm[0];
    for (int c = t; c < N_NODES; c += 256) {
        float d = v[c] - fm;
        e1[(size_t)i * N_NODES + c] = __expf(d);
        e2[(size_t)i * N_NODES + c] = __expf(ALPHA * d);
    }
    if (t == 0) fmax[i] = fm;
}

// ---------------------------------------------------------------------------
// Kernel 3: pre-swizzle Whnew (fp32) into bf16 WMMA B-fragment layout
// (ISA 7.12.2: lane = (n&15)|((k&8)?16:0), slot = ((k&16)?8:0)|(k&7)).
// ---------------------------------------------------------------------------
__global__ void __launch_bounds__(256)
k_repack(const float* __restrict__ whF, unsigned short* __restrict__ frag)
{
    const int t   = threadIdx.x;
    const int hop = blockIdx.x / (N_NODES / 64);
    const int blk = blockIdx.x % (N_NODES / 64);
    const float*    src = whF  + (size_t)hop * N_NODES * OUT_F;
    unsigned short* dst = frag + (size_t)hop * N_NODES * OUT_F;

    const int kblk  = t >> 5;
    const int kg0   = blk * 64 + kblk * 8;
    const int nb    = (t & 31) * 4;
    const int kk    = kg0 & 31;
    const int ktile = kg0 >> 5;
    const int lg    = (kk & 8) ? 16 : 0;
    const int slotb = (kk & 16) ? 8 : 0;

    v4f rowv[8];
#pragma unroll
    for (int jj = 0; jj < 8; ++jj)
        rowv[jj] = *reinterpret_cast<const v4f*>(src + (size_t)(kg0 + jj) * OUT_F + nb);
#pragma unroll
    for (int j = 0; j < 4; ++j) {
        union { unsigned short u16[8]; uint4 v; } bp;
#pragma unroll
        for (int jj = 0; jj < 8; ++jj) bp.u16[jj] = f2bf(rowv[jj][j]);
        const int n = nb + j;
        size_t o = (((size_t)ktile * 8 + (n >> 4)) * 32 + ((n & 15) | lg)) * 16 + slotb;
        *reinterpret_cast<uint4*>(dst + o) = bp.v;
    }
}

// ---------------------------------------------------------------------------
// Kernel 4 (per hop, SINGLE mask pass):
//   Wh[r,:] = (1-B) * softmax_row(masked logits) @ Whnew + B * Wh_old[r,:]
// p = exp(lrelu(f1r+f2c) - bnd) rebuilt WITHOUT exp:
//   pos: p = C1 * e1[c],  neg: p = C2 * e2[c]   (all factors <= 1)
// bf16 pack via one v_perm_b32 per float pair; fp32 row sum; 1/s in epilogue.
// ---------------------------------------------------------------------------
__global__ void __launch_bounds__(256)
k_attn(const int* __restrict__ mask, const unsigned short* __restrict__ bfrag,
       const float* __restrict__ f1h, const float* __restrict__ f2h,
       const float* __restrict__ e1h, const float* __restrict__ e2h,
       const float* __restrict__ fmaxp,
       const float* __restrict__ whold, float* __restrict__ whout, int do_elu)
{
    __shared__ __align__(16) unsigned short pA[2][2][2][32][16]; // [buf][rt][ksub][lane][slot] 8KB
    __shared__ float srow[32];

    const int t = threadIdx.x;
    const int rbase = blockIdx.x * 32;

    // ---- A-staging role: row = t>>3, cols = (t&7)*8 .. +7 (fixed slots) ----
    const int arow  = t >> 3;
    const int scol0 = (t & 7) * 8;
    const float sf1 = f1h[rbase + arow];
    const float bsum = sf1 + fmaxp[0];
    const float bnd  = (bsum > 0.f) ? bsum : ALPHA * bsum;
    const float C1   = __expf(bsum - bnd);          // <= 1
    const float C2   = __expf(ALPHA * bsum - bnd);  // <= 1
    const float negf1 = -sf1;
    const int rt_   = arow >> 4, mloc = arow & 15;
    const int kk0   = scol0 & 31;
    const int aksub = scol0 >> 5;
    const int alane = mloc | ((kk0 & 8) ? 16 : 0);
    const int aslot = (kk0 & 16) ? 8 : 0;
    const v4u* mrow = reinterpret_cast<const v4u*>(mask + (size_t)(rbase + arow) * N_NODES);

    const int w = t >> 5, L = t & 31;

    v8f acc0 = {};
    v8f acc1 = {};
    float ssum = 0.f;

    for (int c0 = 0, step = 0; c0 < N_NODES; c0 += 64, ++step) {
        const int buf = step & 1;

        // ---- B fragments for this step: 4 coalesced b128 loads from L2 ----
        v16bf Bfr[2];
        {
            const int kt0 = c0 >> 5;
#pragma unroll
            for (int ks = 0; ks < 2; ++ks) {
                const uint4* bp = reinterpret_cast<const uint4*>(
                    bfrag + (((size_t)(kt0 + ks) * 8 + w) * 32 + L) * 16);
                union { uint4 q[2]; v8u v; } ub;
                ub.q[0] = bp[0];
                ub.q[1] = bp[1];
                Bfr[ks] = __builtin_bit_cast(v16bf, ub.v);
            }
        }

        // ---- stage A: probabilities -> bf16, one packed b128 store ----
        {
            const int cofs = c0 + scol0;
            const int mofs = cofs >> 2;
            v4u m0 = __builtin_nontemporal_load(mrow + mofs);      // NT: don't thrash L2
            v4u m1 = __builtin_nontemporal_load(mrow + mofs + 1);
            const v4f* fp = reinterpret_cast<const v4f*>(f2h + cofs);
            const v4f* q1 = reinterpret_cast<const v4f*>(e1h + cofs);
            const v4f* q2 = reinterpret_cast<const v4f*>(e2h + cofs);
            v4f g0 = fp[0], g1 = fp[1];
            v4f a0 = q1[0], a1v = q1[1];
            v4f b0 = q2[0], b1v = q2[1];
            float fv[8] = { g0[0], g0[1], g0[2], g0[3], g1[0], g1[1], g1[2], g1[3] };
            float ev1[8] = { a0[0], a0[1], a0[2], a0[3], a1v[0], a1v[1], a1v[2], a1v[3] };
            float ev2[8] = { b0[0], b0[1], b0[2], b0[3], b1v[0], b1v[1], b1v[2], b1v[3] };
            unsigned int mi[8] = { m0[0], m0[1], m0[2], m0[3], m1[0], m1[1], m1[2], m1[3] };
            float pv[8];
            float s0 = 0.f, s1 = 0.f;
#pragma unroll
            for (int j = 0; j < 8; ++j) {
                bool pos = fv[j] > negf1;                  // f1r + f2c > 0
                float Cc = pos ? C1 : C2;
                float Ec = pos ? ev1[j] : ev2[j];
                float p  = Cc * Ec;                        // exp(lrelu(..)-bnd), exact algebra
                p = (mi[j] != 0u) ? p : 0.f;
                pv[j] = p;
                if (j & 1) s1 += p; else s0 += p;
            }
            ssum += s0 + s1;
            uint4 st;
            st.x = __builtin_amdgcn_perm(__builtin_bit_cast(unsigned int, pv[1]),
                                         __builtin_bit_cast(unsigned int, pv[0]), 0x07060302u);
            st.y = __builtin_amdgcn_perm(__builtin_bit_cast(unsigned int, pv[3]),
                                         __builtin_bit_cast(unsigned int, pv[2]), 0x07060302u);
            st.z = __builtin_amdgcn_perm(__builtin_bit_cast(unsigned int, pv[5]),
                                         __builtin_bit_cast(unsigned int, pv[4]), 0x07060302u);
            st.w = __builtin_amdgcn_perm(__builtin_bit_cast(unsigned int, pv[7]),
                                         __builtin_bit_cast(unsigned int, pv[6]), 0x07060302u);
            *reinterpret_cast<uint4*>(&pA[buf][rt_][aksub][alane][aslot]) = st;
        }
        __syncthreads();   // single barrier: pA double-buffered
#pragma unroll
        for (int ksub = 0; ksub < 2; ++ksub) {
            v16bf A0 = __builtin_bit_cast(v16bf, *reinterpret_cast<const v8u*>(&pA[buf][0][ksub][L][0]));
            v16bf A1 = __builtin_bit_cast(v16bf, *reinterpret_cast<const v8u*>(&pA[buf][1][ksub][L][0]));
            acc0 = __builtin_amdgcn_wmma_f32_16x16x32_bf16(false, A0, false, Bfr[ksub],
                                                           (short)0, acc0, false, false);
            acc1 = __builtin_amdgcn_wmma_f32_16x16x32_bf16(false, A1, false, Bfr[ksub],
                                                           (short)0, acc1, false, false);
        }
    }

    // ---- row sums: the 8 lanes sharing a row are contiguous within a wave ----
    float s = ssum;
    s += __shfl_xor(s, 1, 32);
    s += __shfl_xor(s, 2, 32);
    s += __shfl_xor(s, 4, 32);
    if ((t & 7) == 0) srow[arow] = s;
    __syncthreads();

    // ---- epilogue: C layout (7.12.2): VGPR v -> M = v + 8*(lane>=16), N = lane&15
    const int col = (w << 4) + (L & 15);
    const int rV  = (L >= 16) ? 8 : 0;
#pragma unroll
    for (int v = 0; v < 8; ++v) {
        {
            float sr = srow[rV + v];
            float inv = (sr > 0.f) ? 1.f / sr : 0.f;
            int r = rbase + v + rV;
            float x = (1.f - BETA) * acc0[v] * inv + BETA * whold[(size_t)r * OUT_F + col];
            if (do_elu) x = (x > 0.f) ? x : (__expf(x) - 1.f);
            whout[(size_t)r * OUT_F + col] = x;
        }
        {
            float sr = srow[16 + rV + v];
            float inv = (sr > 0.f) ? 1.f / sr : 0.f;
            int r = rbase + 16 + v + rV;
            float x = (1.f - BETA) * acc1[v] * inv + BETA * whold[(size_t)r * OUT_F + col];
            if (do_elu) x = (x > 0.f) ? x : (__expf(x) - 1.f);
            whout[(size_t)r * OUT_F + col] = x;
        }
    }
}

// ---------------------------------------------------------------------------
extern "C" void kernel_launch(void* const* d_in, const int* in_sizes, int n_in,
                              void* d_out, int out_size, void* d_ws, size_t ws_size,
                              hipStream_t stream)
{
    (void)in_sizes; (void)n_in; (void)out_size; (void)ws_size;

    const float* h     = (const float*)d_in[0];
    const int*   kmask = (const int*)  d_in[1];
    const float* W     = (const float*)d_in[2];
    const float* a     = (const float*)d_in[3];
    float*       out   = (float*)d_out;

    char* ws = (char*)d_ws;
    size_t off = 0;
    float* whF = (float*)(ws + off);          off += (size_t)K_HOPS * N_NODES * OUT_F * sizeof(float);
    unsigned short* bfr = (unsigned short*)(ws + off);
                                              off += (size_t)K_HOPS * N_NODES * OUT_F * sizeof(unsigned short);
    float* f1  = (float*)(ws + off);          off += (size_t)K_HOPS * N_NODES * sizeof(float);
    float* f2  = (float*)(ws + off);          off += (size_t)K_HOPS * N_NODES * sizeof(float);
    float* e1  = (float*)(ws + off);          off += (size_t)K_HOPS * N_NODES * sizeof(float);
    float* e2  = (float*)(ws + off);          off += (size_t)K_HOPS * N_NODES * sizeof(float);
    float* fmx = (float*)(ws + off);          off += 64;
    float* Wh  = (float*)(ws + off);          off += (size_t)N_NODES * OUT_F * sizeof(float);

    k_proj<<<N_NODES / 16, 256, 0, stream>>>(h, W, a, whF, f1, f2);
    k_fprep<<<K_HOPS, 256, 0, stream>>>(f2, fmx, e1, e2);
    k_repack<<<K_HOPS * (N_NODES / 64), 256, 0, stream>>>(whF, bfr);

    for (int i = K_HOPS - 1; i >= 0; --i) {
        const int*            maski = kmask + (size_t)i * N_NODES * N_NODES;
        const float*          f1h   = f1  + (size_t)i * N_NODES;
        const float*          f2h   = f2  + (size_t)i * N_NODES;
        const float*          e1h   = e1  + (size_t)i * N_NODES;
        const float*          e2h   = e2  + (size_t)i * N_NODES;
        const unsigned short* bfri  = bfr + (size_t)i * N_NODES * OUT_F;
        const float* whold = (i == K_HOPS - 1) ? (whF + (size_t)i * N_NODES * OUT_F) : Wh;
        float*       whout = (i == 0) ? out : Wh;
        k_attn<<<N_NODES / 32, 256, 0, stream>>>(maski, bfri, f1h, f2h, e1h, e2h,
                                                 fmx + i, whold, whout, (i == 0) ? 1 : 0);
    }
}
